// EngramLayer_56710748176506
// MI455X (gfx1250) — matrix operand order repaired and checked by reading, not compiled, not run on hardware
//
#include <hip/hip_runtime.h>
#include <hip/hip_bf16.h>

#define HC      4
#define S_LEN   2048
#define B_SZ    4
#define NTOK    (B_SZ * S_LEN)      // 8192
#define EDIM    512
#define HID     1024
#define NOUT    (HID * (1 + HC))    // 5120
#define KERN    4
#define DIL     3
#define LDS_ROW 528                 // 512 bf16 + 16 pad -> 1056 B row stride

typedef __bf16 bf16;
typedef __attribute__((ext_vector_type(16))) __bf16 v16bf;
typedef __attribute__((ext_vector_type(8)))  float  v8f;
typedef __attribute__((ext_vector_type(4)))  unsigned int u32x4;
typedef __attribute__((ext_vector_type(4)))  unsigned int uint32x4;
typedef __attribute__((ext_vector_type(4)))  int          int32x4;
typedef __attribute__((ext_vector_type(8)))  int          int32x8;

union Frag { u32x4 q[2]; v16bf v; };

#if defined(__gfx1250__) && __has_builtin(__builtin_amdgcn_tensor_load_to_lds)
#define HAVE_TDM 1
#endif

// ---------------------------------------------------------------------------
// 1) convert concatenated [w_v ; w_k] (5120 x 512 f32) to bf16
// ---------------------------------------------------------------------------
__global__ void __launch_bounds__(256)
convert_w_kernel(const float* __restrict__ wv, const float* __restrict__ wk,
                 bf16* __restrict__ W) {
    int i = blockIdx.x * 1024 + threadIdx.x;
#pragma unroll
    for (int r = 0; r < 4; ++r, i += 256) {
        float x = (i < HID * EDIM) ? wv[i] : wk[i - HID * EDIM];
        W[i] = (bf16)x;
    }
}

// ---------------------------------------------------------------------------
// 2) hashed multi-head embedding gather -> bf16 A matrix [NTOK x 512]
// ---------------------------------------------------------------------------
__global__ void __launch_bounds__(256)
gather_kernel(const int* __restrict__ hidx, const int* __restrict__ offs,
              const float* __restrict__ table, bf16* __restrict__ emb) {
    const int t = blockIdx.x;
#pragma unroll
    for (int r = 0; r < 2; ++r) {
        int e = threadIdx.x + r * 256;
        int head = e >> 6, d = e & 63;
        long row = (long)hidx[t * 8 + head] + (long)offs[head];
        emb[(long)t * EDIM + e] = (bf16)table[row * 64 + d];
    }
}

// ---------------------------------------------------------------------------
// 3) fused GEMM: C[8192 x 5120] = A[8192 x 512] * W^T, bf16 in / f32 out.
//    Block = 8 waves (4 along M x 2 along N) -> 128x128 tile;
//    wave tile 32x64 = 2x4 WMMA 16x16x32 accumulating f32.
//    A block tile (128 x 512 bf16, 128 KB) staged into LDS by one TDM
//    tensor_load_to_lds with D# padding (1024B rows + 32B pad) to spread
//    LDS banks; fragments then come from ds_load_b128.
// ---------------------------------------------------------------------------
__global__ void __launch_bounds__(256)
engram_gemm_kernel(const bf16* __restrict__ A, const bf16* __restrict__ W,
                   float* __restrict__ value, float* __restrict__ keys) {
    extern __shared__ bf16 shA[];          // 128 * LDS_ROW bf16 = 135168 B

    const int lane = threadIdx.x & 31;
    const int wave = threadIdx.x >> 5;
    const int wm = wave & 3;               // 4 waves along M
    const int wn = wave >> 2;              // 2 waves along N
    const int mBase = blockIdx.x * 128 + wm * 32;
    const int nBase = blockIdx.y * 128 + wn * 64;
    const int half = lane >> 4;
    const int l16  = lane & 15;

    // ---- stage A tile: global -> LDS -------------------------------------
#ifdef HAVE_TDM
    if (threadIdx.x < 32) {                // wave 0 issues one TDM descriptor
        const unsigned lds_off = (unsigned)(size_t)shA;
        const unsigned long long ga =
            (unsigned long long)(const void*)(A + (long)blockIdx.x * 128 * EDIM);
        // D# group0: count=1 | lds_addr | global_addr[56:0] | type=2
        uint32x4 g0 = { 1u,
                        lds_off,
                        (unsigned)(ga & 0xffffffffu),
                        (unsigned)((ga >> 32) & 0x01ffffffu) | (2u << 30) };
        // D# group1: data_size=2B, pad_enable, pad_interval=256DW, pad_amount=8DW,
        //            tensor_dim0=512, tensor_dim1=8192, tile 512x128, stride0=512
        int32x8 g1 = { (int)((1u << 16) | (1u << 20) | (7u << 22) | (7u << 25)),
                       (int)(512u << 16),     // tensor_dim0[15:0]
                       (int)(8192u << 16),    // tensor_dim1[15:0]
                       (int)(512u << 16),     // tile_dim0
                       128,                   // tile_dim1
                       512,                   // tensor_dim0_stride[31:0]
                       0, 0 };
        int32x4 zz = {0, 0, 0, 0};
#if __clang_major__ >= 23
        int32x8 z8 = {0, 0, 0, 0, 0, 0, 0, 0};
        __builtin_amdgcn_tensor_load_to_lds(g0, g1, zz, zz, z8, 0);
#else
        __builtin_amdgcn_tensor_load_to_lds(g0, g1, zz, zz, 0);
#endif
        __builtin_amdgcn_s_wait_tensorcnt(0);
    }
    __syncthreads();
#else
    {
        const bf16* gsrc = A + (long)blockIdx.x * 128 * EDIM;
#pragma unroll
        for (int r = 0; r < 32; ++r) {
            int e = threadIdx.x + r * 256;          // 8192 16-elem chunks
            int row = e >> 5, c = (e & 31) * 16;
            *(u32x4*)(shA + row * LDS_ROW + c) = *(const u32x4*)(gsrc + (long)row * EDIM + c);
        }
        __syncthreads();
    }
#endif

    // ---- accumulators -----------------------------------------------------
    const v8f vzero = {0.f,0.f,0.f,0.f,0.f,0.f,0.f,0.f};
    v8f acc[2][4];
#pragma unroll
    for (int i = 0; i < 2; ++i)
#pragma unroll
        for (int j = 0; j < 4; ++j) acc[i][j] = vzero;

    // A fragments from LDS (documented 16x32 bf16 layout):
    //   lane: M=l%16, half=l/16; elems i<8 -> k=half*8+i ; i>=8 -> k+16
    const bf16* la0 = shA + (wm * 32 +      l16) * LDS_ROW + half * 8;
    const bf16* la1 = shA + (wm * 32 + 16 + l16) * LDS_ROW + half * 8;
    // B fragments from global (32x16 layout): lane N=l%16, elems i<8 -> k=half*8+i ; i>=8 -> k+16
    const bf16* pb0 = W + (long)(nBase + l16) * EDIM + half * 8;

    for (int k0 = 0; k0 < EDIM; k0 += 32) {
        Frag a0, a1, b[4];
        a0.q[0] = *(const u32x4*)(la0 + k0);
        a0.q[1] = *(const u32x4*)(la0 + k0 + 16);
        a1.q[0] = *(const u32x4*)(la1 + k0);
        a1.q[1] = *(const u32x4*)(la1 + k0 + 16);
#pragma unroll
        for (int j = 0; j < 4; ++j) {
            const bf16* p = pb0 + (long)j * 16 * EDIM + k0;
            b[j].q[0] = *(const u32x4*)(p);
            b[j].q[1] = *(const u32x4*)(p + 16);
        }
        __builtin_prefetch(pb0 + k0 + 64, 0, 0);
#pragma unroll
        for (int j = 0; j < 4; ++j) {
            acc[0][j] = __builtin_amdgcn_wmma_f32_16x16x32_bf16(
                false, a0.v, false, b[j].v, (short)0, acc[0][j], false, false);
            acc[1][j] = __builtin_amdgcn_wmma_f32_16x16x32_bf16(
                false, a1.v, false, b[j].v, (short)0, acc[1][j], false, false);
        }
    }

    // ---- epilogue: block-uniform dest base + stride (no per-store select) -
    // Blocks y<8 are entirely `value`, y>=8 entirely `keys` (m = (y-8)/8).
    float* obase;
    long ostride;
    if (blockIdx.y < 8) {
        obase = value + blockIdx.y * 128;
        ostride = HID;
    } else {
        const int m = (blockIdx.y - 8) >> 3;
        obase = keys + (long)m * HID + ((blockIdx.y * 128) & (HID - 1));
        ostride = (long)HC * HID;
    }
    float* wbase = obase + wn * 64 + l16;
#pragma unroll
    for (int i = 0; i < 2; ++i) {
#pragma unroll
        for (int j = 0; j < 4; ++j) {
#pragma unroll
            for (int r = 0; r < 8; ++r) {
                const int grow = mBase + i * 16 + r + half * 8;
                wbase[(long)grow * ostride + j * 16] = acc[i][j][r];
            }
        }
    }
}

// ---------------------------------------------------------------------------
// 4) gate: per (token,m) row — RMSNorm dot gate, gated = sigmoid(g)*value,
//    written in-place over keys; also store inv-RMS for the ShortConv norm.
// ---------------------------------------------------------------------------
__global__ void __launch_bounds__(256)
gate_kernel(const float* __restrict__ hidden, const float* __restrict__ value,
            float* __restrict__ kg, const float* __restrict__ nhw,
            const float* __restrict__ nkw, float* __restrict__ inv) {
    const int bi = blockIdx.x;                // t*HC + m
    const int t = bi >> 2, m = bi & 3;
    const long rbase = (long)bi * HID;

    float k[4], h[4], v[4];
    float skk = 0.f, shh = 0.f, skh = 0.f, svv = 0.f;
#pragma unroll
    for (int r = 0; r < 4; ++r) {
        const int hc = threadIdx.x + r * 256;
        k[r] = kg[rbase + hc];
        h[r] = hidden[rbase + hc];
        v[r] = value[(long)t * HID + hc];
        skk += k[r] * k[r];
        shh += h[r] * h[r];
        svv += v[r] * v[r];
        skh += (k[r] * nkw[m * HID + hc]) * (h[r] * nhw[m * HID + hc]);
    }
#pragma unroll
    for (int o = 16; o > 0; o >>= 1) {
        skk += __shfl_xor(skk, o, 32);
        shh += __shfl_xor(shh, o, 32);
        skh += __shfl_xor(skh, o, 32);
        svv += __shfl_xor(svv, o, 32);
    }
    __shared__ float red[4][8];
    const int wave = threadIdx.x >> 5;
    if ((threadIdx.x & 31) == 0) {
        red[0][wave] = skk; red[1][wave] = shh; red[2][wave] = skh; red[3][wave] = svv;
    }
    __syncthreads();
    skk = shh = skh = svv = 0.f;
#pragma unroll
    for (int w = 0; w < 8; ++w) {
        skk += red[0][w]; shh += red[1][w]; skh += red[2][w]; svv += red[3][w];
    }

    const float GEPS = 1.1920929e-07f;               // FLT_EPSILON
    const float rk = rsqrtf(skk * (1.f / HID) + GEPS);
    const float rh = rsqrtf(shh * (1.f / HID) + GEPS);
    const float pre = skh * rk * rh * (1.f / 32.f);  // / sqrt(1024)
    const float sg = (pre > 0.f) ? 1.f : ((pre < 0.f) ? -1.f : 0.f);
    const float g0 = sqrtf(fmaxf(fabsf(pre), 1e-6f)) * sg;
    const float g  = 1.f / (1.f + expf(-g0));

    if (threadIdx.x == 0)
        inv[bi] = rsqrtf(g * g * svv * (1.f / HID) + 1e-5f);  // inv rms(g*v)

#pragma unroll
    for (int r = 0; r < 4; ++r) {
        const int hc = threadIdx.x + r * 256;
        kg[rbase + hc] = g * v[r];                    // gated, in-place
    }
}

// ---------------------------------------------------------------------------
// 5) ShortConv: out = SiLU(depthwise dilated causal conv(RMSNorm(gated)))
//                + gated + hidden
// ---------------------------------------------------------------------------
__global__ void __launch_bounds__(256)
conv_kernel(const float* __restrict__ gated, const float* __restrict__ inv,
            const float* __restrict__ hidden, const float* __restrict__ scw,
            const float* __restrict__ cw, float* __restrict__ out) {
    const int bi = blockIdx.x;               // t*HC + m
    const int t = bi >> 2, m = bi & 3;
    const int s = t & (S_LEN - 1);
#pragma unroll
    for (int r = 0; r < 4; ++r) {
        const int hc = threadIdx.x + r * 256;
        const long cur = (long)bi * HID + hc;
        float a = 0.f;
#pragma unroll
        for (int j = 0; j < 4; ++j) {
            const int d = (3 - j) * DIL;     // tap j reads x[s - (3-j)*dil]
            if (s - d >= 0) {
                const long src = ((long)(t - d) * HC + m) * HID + hc;
                const float xn = gated[src] * inv[(t - d) * HC + m] * scw[m * HID + hc];
                a += cw[(m * HID + hc) * KERN + j] * xn;
            }
        }
        const float so = a / (1.f + expf(-a));       // SiLU
        out[cur] = so + gated[cur] + hidden[cur];
    }
}

// ---------------------------------------------------------------------------
extern "C" void kernel_launch(void* const* d_in, const int* in_sizes, int n_in,
                              void* d_out, int out_size, void* d_ws, size_t ws_size,
                              hipStream_t stream) {
    const int*   hidx   = (const int*)  d_in[0];
    const float* hidden = (const float*)d_in[1];
    const float* table  = (const float*)d_in[2];
    const int*   offs   = (const int*)  d_in[3];
    const float* wv     = (const float*)d_in[4];
    const float* wk     = (const float*)d_in[5];
    const float* nhw    = (const float*)d_in[6];
    const float* nkw    = (const float*)d_in[7];
    const float* scw    = (const float*)d_in[8];
    const float* cw     = (const float*)d_in[9];
    float* out = (float*)d_out;

    char* ws = (char*)d_ws;
    // workspace layout (bytes)
    bf16*  emb   = (bf16*) (ws);                         //  8,388,608
    bf16*  W     = (bf16*) (ws +  8388608);              //  5,242,880
    float* value = (float*)(ws + 13631488);              // 33,554,432
    float* kg    = (float*)(ws + 47185920);              // 134,217,728 (keys -> gated)
    float* inv   = (float*)(ws + 181403648);             //    131,072

    convert_w_kernel<<<2560, 256, 0, stream>>>(wv, wk, W);
    gather_kernel<<<NTOK, 256, 0, stream>>>(hidx, offs, table, emb);
    engram_gemm_kernel<<<dim3(NTOK / 128, NOUT / 128), 256,
                         128 * LDS_ROW * sizeof(bf16), stream>>>(emb, W, value, kg);
    gate_kernel<<<NTOK * HC, 256, 0, stream>>>(hidden, value, kg, nhw, nkw, inv);
    conv_kernel<<<NTOK * HC, 256, 0, stream>>>(kg, inv, hidden, scw, cw, out);
}
